// EfficientAttention_9165460210187
// MI455X (gfx1250) — matrix operand order repaired and checked by reading, not compiled, not run on hardware
//
#include <hip/hip_runtime.h>

typedef __bf16 bf16_t;
typedef __attribute__((ext_vector_type(16))) __bf16 v16bf;
typedef __attribute__((ext_vector_type(8)))  float  v8f;
typedef int v4i __attribute__((vector_size(16)));
typedef __attribute__((address_space(1))) v4i g_v4i;   // global int4
typedef __attribute__((address_space(3))) v4i l_v4i;   // LDS int4

#define B_     8
#define C_     256
#define N_     16384   // 128*128
#define KC_    256
#define VC_    256
#define HEADS_ 8
#define DK_    32
#define DV_    32

#if defined(__AMDGCN__) && __has_builtin(__builtin_amdgcn_global_load_async_to_lds_b128) && \
    __has_builtin(__builtin_amdgcn_s_wait_asynccnt)
#define HAVE_ASYNC_LDS 1
#else
#define HAVE_ASYNC_LDS 0
#endif

// ---------------------------------------------------------------------------
// Kernel 1: projection GEMM.  Out[b,m,n] = bf16( sum_k W[m,k]*X[b,k,n] + bias[m] )
// WG: 256 threads = 8 waves; WG tile 64(M) x 256(N); wave tile 32x64
// (2x4 WMMA 16x16x32 bf16 frags).  A tile row-major in LDS (contiguous frag
// reads); B tile stored TRANSPOSED [n][k] so B-frag reads are 2x ds_load_b128.
// ---------------------------------------------------------------------------
__global__ __launch_bounds__(256) void gemm_proj_kernel(
    const float* __restrict__ W,     // [256, 256]
    const float* __restrict__ bias,  // [256]
    const float* __restrict__ X,     // [B, 256, N]
    bf16_t* __restrict__ Out)        // [B, 256, N] bf16
{
  __shared__ bf16_t As[64][40];     // [m][k]
  __shared__ bf16_t Bst[256][40];   // [n][k]  (transposed B tile)
  const int b   = blockIdx.z;
  const int m0  = blockIdx.y * 64;
  const int n0  = blockIdx.x * 256;
  const int tid = threadIdx.x;
  const int wave = tid >> 5;
  const int lane = tid & 31;
  const int lo = lane & 15, hi = lane >> 4;
  const int wm = wave >> 2;   // 0..1
  const int wn = wave & 3;    // 0..3
  const float* Xb = X + (size_t)b * C_ * N_;

  v8f acc[2][4] = {};

  const int akr = tid >> 2;          // A stage: row 0..63
  const int akg = (tid & 3) * 8;     //          k group
  const int bkr = tid >> 3;          // B stage: k row 0..31
  const int bng = (tid & 7) * 32;    //          n group (32 cols)

  for (int k0 = 0; k0 < C_; k0 += 32) {
    if (k0 + 32 < C_) {  // prefetch next K tile (global_prefetch_b8)
      __builtin_prefetch(&W[(m0 + akr) * C_ + k0 + 32 + akg], 0, 1);
      __builtin_prefetch(&Xb[(size_t)(k0 + 32 + bkr) * N_ + n0 + bng], 0, 1);
    }
    #pragma unroll
    for (int i = 0; i < 8; ++i)
      As[akr][akg + i] = (bf16_t)W[(m0 + akr) * C_ + k0 + akg + i];
    #pragma unroll
    for (int i = 0; i < 32; ++i)   // fp32 load (coalesced) -> bf16 -> transposed store
      Bst[bng + i][bkr] = (bf16_t)Xb[(size_t)(k0 + bkr) * N_ + n0 + bng + i];
    __syncthreads();

    v16bf afrag[2], bfrag[4];
    #pragma unroll
    for (int r = 0; r < 2; ++r) {
      const int row = wm * 32 + r * 16 + lo;
      #pragma unroll
      for (int e = 0; e < 8; ++e) afrag[r][e]     = As[row][e + hi * 8];
      #pragma unroll
      for (int e = 0; e < 8; ++e) afrag[r][8 + e] = As[row][16 + e + hi * 8];
    }
    #pragma unroll
    for (int c = 0; c < 4; ++c) {
      const int col = wn * 64 + c * 16 + lo;
      #pragma unroll
      for (int e = 0; e < 16; ++e) bfrag[c][e] = Bst[col][(e & 7) + hi * 8 + (e >> 3) * 16];
    }
    #pragma unroll
    for (int r = 0; r < 2; ++r)
      #pragma unroll
      for (int c = 0; c < 4; ++c)
        acc[r][c] = __builtin_amdgcn_wmma_f32_16x16x32_bf16(
            false, afrag[r], false, bfrag[c], (short)0, acc[r][c], false, false);
    __syncthreads();
  }

  bf16_t* Ob = Out + (size_t)b * 256 * N_;
  #pragma unroll
  for (int r = 0; r < 2; ++r) {
    #pragma unroll
    for (int c = 0; c < 4; ++c) {
      const int colg = n0 + wn * 64 + c * 16 + lo;
      #pragma unroll
      for (int v = 0; v < 8; ++v) {
        const int rowg = m0 + wm * 32 + r * 16 + hi * 8 + v;
        Ob[(size_t)rowg * N_ + colg] = (bf16_t)(acc[r][c][v] + bias[rowg]);
      }
    }
  }
}

// ---------------------------------------------------------------------------
// Kernel 2: per-row (b, kc) max and sum(exp) over N for key softmax.
// ---------------------------------------------------------------------------
__global__ __launch_bounds__(256) void key_stats_kernel(
    const bf16_t* __restrict__ Keys, float* __restrict__ rowmax,
    float* __restrict__ rowsum)
{
  const int row = blockIdx.x;   // 0 .. B*KC-1
  const bf16_t* kr = Keys + (size_t)row * N_;
  __shared__ float red[256];
  float m = -3.0e38f;
  for (int n = threadIdx.x; n < N_; n += 256) m = fmaxf(m, (float)kr[n]);
  red[threadIdx.x] = m;
  __syncthreads();
  for (int s = 128; s > 0; s >>= 1) {
    if (threadIdx.x < s)
      red[threadIdx.x] = fmaxf(red[threadIdx.x], red[threadIdx.x + s]);
    __syncthreads();
  }
  m = red[0];
  __syncthreads();
  float sum = 0.f;
  for (int n = threadIdx.x; n < N_; n += 256) sum += __expf((float)kr[n] - m);
  red[threadIdx.x] = sum;
  __syncthreads();
  for (int s = 128; s > 0; s >>= 1) {
    if (threadIdx.x < s) red[threadIdx.x] += red[threadIdx.x + s];
    __syncthreads();
  }
  if (threadIdx.x == 0) { rowmax[row] = m; rowsum[row] = red[0]; }
}

// ---------------------------------------------------------------------------
// Kernel 3a: zero the context accumulator.
// ---------------------------------------------------------------------------
__global__ void zero_kernel(float* __restrict__ p, int n)
{
  int i = blockIdx.x * blockDim.x + threadIdx.x;
  if (i < n) p[i] = 0.f;
}

// ---------------------------------------------------------------------------
// Kernel 3b: context[b,h,k,v] += sum_n exp(K[k,n]-max[k]) * V[v,n]
// One wave per (b,h, n-chunk); 2x2 WMMA tiles; fp32 atomics reduce the N split.
// ---------------------------------------------------------------------------
__global__ __launch_bounds__(32) void context_kernel(
    const bf16_t* __restrict__ Keys, const bf16_t* __restrict__ Vals,
    const float* __restrict__ rowmax, float* __restrict__ Ctx)
{
  const int bh = blockIdx.y;              // 0..63
  const int b = bh >> 3, h = bh & 7;
  const int lane = threadIdx.x;
  const int lo = lane & 15, hi = lane >> 4;
  const size_t kbase = ((size_t)b * KC_ + h * DK_) * N_;
  const size_t vbase = ((size_t)b * VC_ + h * DV_) * N_;

  float rmax[2];
  #pragma unroll
  for (int r = 0; r < 2; ++r)
    rmax[r] = rowmax[b * KC_ + h * DK_ + r * 16 + lo];

  v8f acc[2][2] = {};

  const int nbeg = blockIdx.x * (N_ / 32);
  const int nend = nbeg + (N_ / 32);
  for (int n0 = nbeg; n0 < nend; n0 += 32) {
    v16bf afrag[2], bfrag[2];
    #pragma unroll
    for (int r = 0; r < 2; ++r) {
      const bf16_t* kp = Keys + kbase + (size_t)(r * 16 + lo) * N_ + n0;
      #pragma unroll
      for (int e = 0; e < 8; ++e)
        afrag[r][e]     = (bf16_t)__expf((float)kp[e + hi * 8] - rmax[r]);
      #pragma unroll
      for (int e = 0; e < 8; ++e)
        afrag[r][8 + e] = (bf16_t)__expf((float)kp[16 + e + hi * 8] - rmax[r]);
    }
    #pragma unroll
    for (int c = 0; c < 2; ++c) {
      const bf16_t* vp = Vals + vbase + (size_t)(c * 16 + lo) * N_ + n0;
      #pragma unroll
      for (int e = 0; e < 16; ++e) bfrag[c][e] = vp[(e & 7) + hi * 8 + (e >> 3) * 16];
    }
    #pragma unroll
    for (int r = 0; r < 2; ++r)
      #pragma unroll
      for (int c = 0; c < 2; ++c)
        acc[r][c] = __builtin_amdgcn_wmma_f32_16x16x32_bf16(
            false, afrag[r], false, bfrag[c], (short)0, acc[r][c], false, false);
  }

  float* Cb = Ctx + (size_t)bh * DK_ * DV_;
  #pragma unroll
  for (int r = 0; r < 2; ++r)
    #pragma unroll
    for (int c = 0; c < 2; ++c)
      #pragma unroll
      for (int v = 0; v < 8; ++v) {
        const int kk = r * 16 + hi * 8 + v;
        const int vv = c * 16 + lo;
        atomicAdd(&Cb[kk * DV_ + vv], acc[r][c][v]);
      }
}

// ---------------------------------------------------------------------------
// Kernel 4: query softmax over the 32 in-head channels (per b,h,n).
// ---------------------------------------------------------------------------
__global__ __launch_bounds__(256) void q_softmax_kernel(
    const bf16_t* __restrict__ Q, bf16_t* __restrict__ Qsm)
{
  const int idx = blockIdx.x * blockDim.x + threadIdx.x;  // over B*HEADS*N
  if (idx >= B_ * HEADS_ * N_) return;
  const int n  = idx & (N_ - 1);
  const int bh = idx / N_;
  const int b = bh / HEADS_, h = bh % HEADS_;
  const size_t base = ((size_t)b * KC_ + h * DK_) * N_ + n;
  float m = -3.0e38f;
  for (int c = 0; c < DK_; ++c)
    m = fmaxf(m, (float)Q[base + (size_t)c * N_]);
  float s = 0.f;
  for (int c = 0; c < DK_; ++c)
    s += __expf((float)Q[base + (size_t)c * N_] - m);
  const float inv = 1.f / s;
  for (int c = 0; c < DK_; ++c)
    Qsm[base + (size_t)c * N_] =
        (bf16_t)(__expf((float)Q[base + (size_t)c * N_] - m) * inv);
}

// ---------------------------------------------------------------------------
// Kernel 5: M2[b, c, h*32+k] = sum_v Wr[c, h*32+v] * Ctx[b,h,k,v] / rowsum[b,hk]
// ---------------------------------------------------------------------------
__global__ __launch_bounds__(256) void m2_kernel(
    const float* __restrict__ Wr, const float* __restrict__ Ctx,
    const float* __restrict__ rowsum, float* __restrict__ M2)
{
  const int idx = blockIdx.x * blockDim.x + threadIdx.x;  // over B*C*256
  if (idx >= B_ * C_ * 256) return;
  const int hk = idx & 255;
  const int c  = (idx >> 8) % C_;
  const int b  = idx / (256 * C_);
  const int h = hk >> 5, k = hk & 31;
  const float* cb = Ctx + (((size_t)b * HEADS_ + h) * DK_ + k) * DV_;
  const float* wr = Wr + (size_t)c * VC_ + h * DV_;
  float s = 0.f;
  #pragma unroll
  for (int v = 0; v < DV_; ++v) s += wr[v] * cb[v];
  M2[idx] = s / rowsum[b * KC_ + h * DK_ + k];
}

// ---------------------------------------------------------------------------
// Kernel 6: final GEMM + bias + residual.
// Out[b,c,n] = sum_j M2[b,c,j] * Qsm[b,j,n] + br[c] + x[b,c,n]   (fp32 out)
// B operand (Qsm) is bf16 in global: staged with async global->LDS copies
// when available (global_load_async_to_lds_b128 + s_wait_asynccnt), then
// LDS->LDS transposed into the fragment-friendly layout.
// ---------------------------------------------------------------------------
__global__ __launch_bounds__(256) void gemm_out_kernel(
    const float* __restrict__ M2, const bf16_t* __restrict__ Qsm,
    const float* __restrict__ br, const float* __restrict__ X,
    float* __restrict__ Out)
{
  __shared__ bf16_t As[64][40];
  __shared__ bf16_t Bst[256][40];
#if HAVE_ASYNC_LDS
  __shared__ bf16_t Qraw[32][256];   // row-major async landing buffer
#endif
  const int b   = blockIdx.z;
  const int m0  = blockIdx.y * 64;
  const int n0  = blockIdx.x * 256;
  const int tid = threadIdx.x;
  const int wave = tid >> 5;
  const int lane = tid & 31;
  const int lo = lane & 15, hi = lane >> 4;
  const int wm = wave >> 2;
  const int wn = wave & 3;
  const float*  Ab = M2 + (size_t)b * C_ * 256;
  const bf16_t* Bb = Qsm + (size_t)b * 256 * N_;

  v8f acc[2][4] = {};

  const int akr = tid >> 2;
  const int akg = (tid & 3) * 8;
  const int bkr = tid >> 3;
  const int bng = (tid & 7) * 32;

  for (int k0 = 0; k0 < 256; k0 += 32) {
    if (k0 + 32 < 256) {
      __builtin_prefetch(&Ab[(m0 + akr) * 256 + k0 + 32 + akg], 0, 1);
      __builtin_prefetch(&Bb[(size_t)(k0 + 32 + bkr) * N_ + n0 + bng], 0, 1);
    }
    #pragma unroll
    for (int i = 0; i < 8; ++i)
      As[akr][akg + i] = (bf16_t)Ab[(m0 + akr) * 256 + k0 + akg + i];

#if HAVE_ASYNC_LDS
    // Async stage: 32x256 bf16 tile = 1024 x 16B chunks, 4 per thread.
    #pragma unroll
    for (int j = 0; j < 4; ++j) {
      const int flat = (tid * 4 + j) * 8;     // bf16 element index
      const int kr = flat >> 8;
      const int nc = flat & 255;
      __builtin_amdgcn_global_load_async_to_lds_b128(
          (g_v4i*)&Bb[(size_t)(k0 + kr) * N_ + n0 + nc],
          (l_v4i*)&Qraw[kr][nc],
          0, 0);
    }
    __builtin_amdgcn_s_wait_asynccnt(0);
    __syncthreads();
    // LDS->LDS transpose into fragment layout.
    #pragma unroll
    for (int i = 0; i < 32; ++i)
      Bst[bng + i][bkr] = Qraw[bkr][bng + i];
#else
    #pragma unroll
    for (int i = 0; i < 32; ++i)
      Bst[bng + i][bkr] = Bb[(size_t)(k0 + bkr) * N_ + n0 + bng + i];
#endif
    __syncthreads();

    v16bf afrag[2], bfrag[4];
    #pragma unroll
    for (int r = 0; r < 2; ++r) {
      const int row = wm * 32 + r * 16 + lo;
      #pragma unroll
      for (int e = 0; e < 8; ++e) afrag[r][e]     = As[row][e + hi * 8];
      #pragma unroll
      for (int e = 0; e < 8; ++e) afrag[r][8 + e] = As[row][16 + e + hi * 8];
    }
    #pragma unroll
    for (int c = 0; c < 4; ++c) {
      const int col = wn * 64 + c * 16 + lo;
      #pragma unroll
      for (int e = 0; e < 16; ++e) bfrag[c][e] = Bst[col][(e & 7) + hi * 8 + (e >> 3) * 16];
    }
    #pragma unroll
    for (int r = 0; r < 2; ++r)
      #pragma unroll
      for (int c = 0; c < 4; ++c)
        acc[r][c] = __builtin_amdgcn_wmma_f32_16x16x32_bf16(
            false, afrag[r], false, bfrag[c], (short)0, acc[r][c], false, false);
    __syncthreads();
  }

  #pragma unroll
  for (int r = 0; r < 2; ++r) {
    #pragma unroll
    for (int c = 0; c < 4; ++c) {
      const int colg = n0 + wn * 64 + c * 16 + lo;
      #pragma unroll
      for (int v = 0; v < 8; ++v) {
        const int rowg = m0 + wm * 32 + r * 16 + hi * 8 + v;
        const size_t o = ((size_t)b * C_ + rowg) * N_ + colg;
        Out[o] = acc[r][c][v] + br[rowg] + X[o];
      }
    }
  }
}

// ---------------------------------------------------------------------------
extern "C" void kernel_launch(void* const* d_in, const int* in_sizes, int n_in,
                              void* d_out, int out_size, void* d_ws, size_t ws_size,
                              hipStream_t stream)
{
  const float* x  = (const float*)d_in[0];
  const float* Wk = (const float*)d_in[1];
  const float* bk = (const float*)d_in[2];
  const float* Wq = (const float*)d_in[3];
  const float* bq = (const float*)d_in[4];
  const float* Wv = (const float*)d_in[5];
  const float* bv = (const float*)d_in[6];
  const float* Wr = (const float*)d_in[7];
  const float* br = (const float*)d_in[8];
  float* out = (float*)d_out;

  char* ws = (char*)d_ws;
  size_t off = 0;
  auto carve = [&](size_t bytes) {
    void* p = ws + off;
    off += (bytes + 255) & ~(size_t)255;
    return p;
  };
  const size_t mapElems = (size_t)B_ * 256 * N_;
  bf16_t* keysB = (bf16_t*)carve(mapElems * sizeof(bf16_t));
  bf16_t* qB    = (bf16_t*)carve(mapElems * sizeof(bf16_t));
  bf16_t* vB    = (bf16_t*)carve(mapElems * sizeof(bf16_t));
  bf16_t* qsmB  = (bf16_t*)carve(mapElems * sizeof(bf16_t));
  float* rowmax = (float*)carve((size_t)B_ * KC_ * sizeof(float));
  float* rowsum = (float*)carve((size_t)B_ * KC_ * sizeof(float));
  float* ctx    = (float*)carve((size_t)B_ * HEADS_ * DK_ * DV_ * sizeof(float));
  float* m2     = (float*)carve((size_t)B_ * C_ * 256 * sizeof(float));

  const dim3 gg(N_ / 256, 256 / 64, B_);
  gemm_proj_kernel<<<gg, 256, 0, stream>>>(Wk, bk, x, keysB);
  gemm_proj_kernel<<<gg, 256, 0, stream>>>(Wq, bq, x, qB);
  gemm_proj_kernel<<<gg, 256, 0, stream>>>(Wv, bv, x, vB);

  key_stats_kernel<<<B_ * KC_, 256, 0, stream>>>(keysB, rowmax, rowsum);

  const int ctxElems = B_ * HEADS_ * DK_ * DV_;
  zero_kernel<<<(ctxElems + 255) / 256, 256, 0, stream>>>(ctx, ctxElems);

  context_kernel<<<dim3(32, B_ * HEADS_), 32, 0, stream>>>(keysB, vB, rowmax, ctx);

  q_softmax_kernel<<<(B_ * HEADS_ * N_) / 256, 256, 0, stream>>>(qB, qsmB);

  m2_kernel<<<(B_ * C_ * 256) / 256, 256, 0, stream>>>(Wr, ctx, rowsum, m2);

  gemm_out_kernel<<<gg, 256, 0, stream>>>(m2, qsmB, br, x, out);
}
